// BDHLayer_17171279249446
// MI455X (gfx1250) — matrix operand order repaired and checked by reading, not compiled, not run on hardware
//
#include <hip/hip_runtime.h>
#include <hip/hip_bf16.h>
#include <math.h>

// Problem constants (match reference)
constexpr int CB  = 2;
constexpr int CNH = 12;
constexpr int CT  = 2048;
constexpr int CD  = 768;
constexpr int CN  = 256;
constexpr int CK  = CNH * CN; // 3072, decoder reduction dim

typedef __bf16 bf16_t;
typedef __attribute__((ext_vector_type(16))) __bf16 v16bf;
typedef __attribute__((ext_vector_type(8)))  float  v8f;

// ---------------------------------------------------------------------------
// WMMA helpers (CDNA5: V_WMMA_F32_16X16X32_BF16, wave32)
// ---------------------------------------------------------------------------
__device__ inline v8f wmma_bf16(v16bf a, v16bf b, v8f c) {
  return __builtin_amdgcn_wmma_f32_16x16x32_bf16(
      false, a, false, b, (short)0, c, false, false);
}

// A operand (16x32) from row-major (m,k) storage, leading dim ld.
// lane L -> row M=L&15; lanes 0-15 hold K {0..7}∪{16..23}, lanes 16-31
// hold K {8..15}∪{24..31}: two contiguous 8-half (16B) chunks per lane.
__device__ inline v16bf load_A_rowmajor(const bf16_t* base, int ld,
                                        int m0, int k0, int lane) {
  const int row = m0 + (lane & 15);
  const int kb  = (lane < 16) ? 0 : 8;
  const bf16_t* p = base + (size_t)row * ld + k0;
  v16bf a;
#pragma unroll
  for (int i = 0; i < 8; ++i) a[i] = p[kb + i];
#pragma unroll
  for (int i = 0; i < 8; ++i) a[8 + i] = p[16 + kb + i];
  return a;
}

// B operand (32x16) where B[k][n] = Mrows[n0+n][k0+k] (Mrows is the
// TRANSPOSED weight copy, so K is contiguous: two 16B runs per lane).
// lane L -> col N=L&15; lanes 0-15 hold K {0..7}∪{16..23}, lanes 16-31 hold K {8..15}∪{24..31}.
__device__ inline v16bf load_B_rowsT(const bf16_t* base, int ld,
                                     int n0, int k0, int lane) {
  const int row = n0 + (lane & 15);
  const int kb  = (lane < 16) ? 0 : 8;
  const bf16_t* p = base + (size_t)row * ld + k0 + kb;
  v16bf b;
#pragma unroll
  for (int i = 0; i < 8; ++i) b[i] = p[i];
#pragma unroll
  for (int i = 0; i < 8; ++i) b[8 + i] = p[16 + i];
  return b;
}

// ---------------------------------------------------------------------------
// Staging kernels
// ---------------------------------------------------------------------------
__global__ void cvt_f32_bf16(const float* __restrict__ in,
                             bf16_t* __restrict__ out, int n) {
  int i = blockIdx.x * blockDim.x + threadIdx.x;
  if (i < n) out[i] = (bf16_t)in[i];
}

// (R,C) f32 row-major -> (C,R) bf16 row-major, one matrix per blockIdx.z.
__global__ __launch_bounds__(256) void transpose_cvt_kernel(
    const float* __restrict__ in, bf16_t* __restrict__ out, int R, int C) {
  __shared__ float tile[32][33];
  const float* src = in + (size_t)blockIdx.z * R * C;
  bf16_t* dst = out + (size_t)blockIdx.z * R * C;
  const int r0 = blockIdx.y * 32, c0 = blockIdx.x * 32;
  const int tr = threadIdx.y, tc = threadIdx.x;
#pragma unroll
  for (int i = 0; i < 32; i += 8)
    tile[tr + i][tc] = src[(size_t)(r0 + tr + i) * C + (c0 + tc)];
  __syncthreads();
#pragma unroll
  for (int i = 0; i < 32; i += 8)
    dst[(size_t)(c0 + tr + i) * R + (r0 + tc)] = (bf16_t)tile[tc][tr + i];
}

// ---------------------------------------------------------------------------
// Kernel A: x_sparse = relu(x @ encoder) per (b,h): (T x D)(D x N)
// block 256 thr = 8 waves arranged 2(M) x 4(N); wave tile 32x32 (4 WMMA/k)
// ---------------------------------------------------------------------------
__global__ __launch_bounds__(256) void encode_gemm_kernel(
    const bf16_t* __restrict__ xbf, const bf16_t* __restrict__ enc_t,
    float* __restrict__ xsp) {
  const int lane = threadIdx.x & 31, wave = threadIdx.x >> 5;
  const int wm = wave & 1, wn = wave >> 1;
  const int t0 = blockIdx.x * 64 + wm * 32;
  const int n0 = blockIdx.y * 128 + wn * 32;
  const int bh = blockIdx.z;
  const int b = bh / CNH, h = bh % CNH;
  const bf16_t* A  = xbf   + (size_t)b * CT * CD;    // T x D
  const bf16_t* Bt = enc_t + (size_t)h * CN * CD;    // N x D (transposed)

  v8f c[2][2];
#pragma unroll
  for (int i = 0; i < 2; ++i)
#pragma unroll
    for (int j = 0; j < 2; ++j)
      c[i][j] = (v8f){0.f,0.f,0.f,0.f,0.f,0.f,0.f,0.f};

  for (int k0 = 0; k0 < CD; k0 += 32) {
    if (k0 + 32 < CD)
      __builtin_prefetch(Bt + (size_t)n0 * CD + k0 + 32, 0, 1);
    v16bf a0 = load_A_rowmajor(A, CD, t0,      k0, lane);
    v16bf a1 = load_A_rowmajor(A, CD, t0 + 16, k0, lane);
    v16bf b0 = load_B_rowsT(Bt, CD, n0,      k0, lane);
    v16bf b1 = load_B_rowsT(Bt, CD, n0 + 16, k0, lane);
    c[0][0] = wmma_bf16(a0, b0, c[0][0]);
    c[0][1] = wmma_bf16(a0, b1, c[0][1]);
    c[1][0] = wmma_bf16(a1, b0, c[1][0]);
    c[1][1] = wmma_bf16(a1, b1, c[1][1]);
  }

  float* out = xsp + (size_t)bh * CT * CN;
#pragma unroll
  for (int mi = 0; mi < 2; ++mi) {
#pragma unroll
    for (int ni = 0; ni < 2; ++ni) {
      const int col = n0 + ni * 16 + (lane & 15);
      const int mb  = t0 + mi * 16 + ((lane >> 4) << 3);
#pragma unroll
      for (int i = 0; i < 8; ++i) {
        float v = c[mi][ni][i];
        out[(size_t)(mb + i) * CN + col] = v > 0.f ? v : 0.f;
      }
    }
  }
}

// ---------------------------------------------------------------------------
// Kernel A2: QR = rope(x_sparse) -> bf16 (QR == KR in reference)
// ---------------------------------------------------------------------------
__global__ void rope_kernel(const float* __restrict__ xsp,
                            bf16_t* __restrict__ qr) {
  const size_t idx = (size_t)blockIdx.x * 256 + threadIdx.x;
  const int n = (int)(idx & (CN - 1));
  const int t = (int)((idx >> 8) & (CT - 1));
  const float q    = (float)(n & ~1);
  const float freq = exp2f(-q * (1.0f / 16.0f)) * 0.15915494309189535f;
  float ph = (float)t * freq;
  ph = (ph - floorf(ph)) * 6.283185307179586f;
  const float v = xsp[idx];
  const float p = xsp[idx ^ 1];
  const float rot = (n & 1) ? p : -p;
  qr[idx] = (bf16_t)(v * cosf(ph) + rot * sinf(ph));
}

// ---------------------------------------------------------------------------
// Kernel B: fused attention middle.
// Per workgroup: one (b,h), one 16-row query tile, 8 waves.
// Key tile = 128: wave w computes the FULL 16x16 score tile for its private
// 16-key slice (8 WMMA over K=256, A operand async-staged in LDS), masks in
// registers, writes bf16 to LDS; then all waves run S(16x128) @ X over their
// 96 yKV columns (24 WMMA). 2 barriers / 128 keys, no LDS atomics.
// ---------------------------------------------------------------------------
__global__ __launch_bounds__(256) void attn_fused_kernel(
    const bf16_t* __restrict__ qr, const bf16_t* __restrict__ xbf_t,
    const float* __restrict__ xsp, const bf16_t* __restrict__ encv_t,
    float* __restrict__ xy_out, bf16_t* __restrict__ flat_bf) {
  __shared__ bf16_t sQ[16][256];       // query tile (A operand), async-staged
  __shared__ bf16_t sSbf[16][128];     // masked scores, bf16 A operand
  __shared__ float  sYKV[16][772];     // yKV tile accumulation target
  __shared__ float  sRed1[16][16];
  __shared__ float  sRed2[16][16];
  __shared__ float  sMean[16];
  __shared__ float  sRstd[16];

  const int lane = threadIdx.x & 31, wave = threadIdx.x >> 5;
  const int t0 = blockIdx.x * 16;
  const int bh = blockIdx.y;
  const int b = bh / CNH, h = bh % CNH;
  const bf16_t* Q   = qr    + (size_t)bh * CT * CN; // T x N (QR == KR)
  const bf16_t* XbT = xbf_t + (size_t)b  * CD * CT; // D x T (transposed V)

  // ---- stage Q tile (16x256 bf16 = 8KB) into LDS with async copies -------
  {
    const int tid = threadIdx.x;
#pragma unroll
    for (int r = 0; r < 2; ++r) {
      const int e   = (tid + r * 256) * 8;      // bf16 element index in tile
      const int row = e >> 8, col = e & 255;
      const bf16_t* g = Q + (size_t)(t0 + row) * CN + col;
      unsigned lds = (unsigned)(size_t)(void*)&sQ[row][col];
      asm volatile("global_load_async_to_lds_b128 %0, %1, off"
                   :: "v"(lds), "v"(g) : "memory");
    }
    asm volatile("s_wait_asynccnt 0x0" ::: "memory");
  }
  __syncthreads();

  v8f yacc[6];
#pragma unroll
  for (int j = 0; j < 6; ++j) yacc[j] = (v8f){0.f,0.f,0.f,0.f,0.f,0.f,0.f,0.f};

  for (int s0 = 0; s0 < t0 + 16; s0 += 128) {
    const int sl0 = s0 + wave * 16;             // this wave's 16-key slice

    // phase 1: full scores for the wave's slice over K=256
    v8f sc = {0.f, 0.f, 0.f, 0.f, 0.f, 0.f, 0.f, 0.f};
    if (sl0 < t0 + 16) {                        // slice not fully masked
      for (int k0 = 0; k0 < CN; k0 += 32) {
        v16bf aq = load_A_rowmajor(&sQ[0][0], CN, 0, k0, lane);
        v16bf bk = load_B_rowsT(Q, CN, sl0, k0, lane);
        sc = wmma_bf16(aq, bk, sc);
      }
    }
    // strict-causal mask in registers, write bf16 slice to LDS
    {
      const int col = lane & 15;
      const int mb  = (lane >> 4) << 3;
#pragma unroll
      for (int i = 0; i < 8; ++i) {
        const int m = mb + i;
        const float v = (sl0 + col < t0 + m) ? sc[i] : 0.f;
        sSbf[m][wave * 16 + col] = (bf16_t)v;
      }
    }
    __syncthreads();

    // phase 2: yKV += S(16x128) @ X[s0:s0+128, wave's 96 cols]
    const int kmax = min(128, t0 + 16 - s0);    // skip all-zero k-steps
    for (int kk = 0; kk < kmax; kk += 32) {
      v16bf as = load_A_rowmajor(&sSbf[0][0], 128, 0, kk, lane);
#pragma unroll
      for (int j = 0; j < 6; ++j) {
        const int d0 = wave * 96 + j * 16;
        v16bf bx = load_B_rowsT(XbT, CT, d0, s0 + kk, lane);
        yacc[j] = wmma_bf16(as, bx, yacc[j]);
      }
    }
    __syncthreads();
  }

  // spill yKV accumulators to LDS
  {
    const int col0 = lane & 15;
    const int mb   = (lane >> 4) << 3;
#pragma unroll
    for (int j = 0; j < 6; ++j) {
      const int cidx = wave * 96 + j * 16 + col0;
#pragma unroll
      for (int i = 0; i < 8; ++i) sYKV[mb + i][cidx] = yacc[j][i];
    }
  }
  __syncthreads();

  // per-row layernorm stats
  {
    const int r = threadIdx.x >> 4, c16 = threadIdx.x & 15;
    float s = 0.f, q = 0.f;
    for (int k = c16; k < CD; k += 16) {
      const float v = sYKV[r][k];
      s += v; q += v * v;
    }
    sRed1[r][c16] = s; sRed2[r][c16] = q;
  }
  __syncthreads();
  if ((threadIdx.x & 15) == 0) {
    const int r = threadIdx.x >> 4;
    float s = 0.f, q = 0.f;
#pragma unroll
    for (int i = 0; i < 16; ++i) { s += sRed1[r][i]; q += sRed2[r][i]; }
    const float m   = s / CD;
    const float var = q / CD - m * m;
    sMean[r] = m;
    sRstd[r] = rsqrtf(var + 1e-5f);
  }
  __syncthreads();

  // y_sparse = relu(yKV_ln @ encoder_v[h]); wave owns 32 output cols
  const bf16_t* EvT = encv_t + (size_t)h * CN * CD;  // N x D (transposed)
  v8f c2[2];
  c2[0] = (v8f){0.f,0.f,0.f,0.f,0.f,0.f,0.f,0.f};
  c2[1] = (v8f){0.f,0.f,0.f,0.f,0.f,0.f,0.f,0.f};
  for (int k0 = 0; k0 < CD; k0 += 32) {
    v16bf a;
    {
      const int row = lane & 15;
      const int kb  = (lane < 16) ? 0 : 8;
      const float m = sMean[row], rs = sRstd[row];
      const float* p = &sYKV[row][k0];
#pragma unroll
      for (int i = 0; i < 8; ++i) a[i]     = (bf16_t)((p[kb + i]      - m) * rs);
#pragma unroll
      for (int i = 0; i < 8; ++i) a[8 + i] = (bf16_t)((p[16 + kb + i] - m) * rs);
    }
#pragma unroll
    for (int ct = 0; ct < 2; ++ct) {
      const int n0 = wave * 32 + ct * 16;
      v16bf bb = load_B_rowsT(EvT, CD, n0, k0, lane);
      c2[ct] = wmma_bf16(a, bb, c2[ct]);
    }
  }

  // epilogue: relu, multiply with x_sparse, write xy (f32) + flat (bf16)
  const float* XS = xsp    + (size_t)bh * CT * CN;
  float*       XY = xy_out + (size_t)bh * CT * CN;
#pragma unroll
  for (int ct = 0; ct < 2; ++ct) {
    const int n  = wave * 32 + ct * 16 + (lane & 15);
    const int mb = t0 + ((lane >> 4) << 3);
#pragma unroll
    for (int i = 0; i < 8; ++i) {
      const int t = mb + i;
      float ys = c2[ct][i];
      ys = ys > 0.f ? ys : 0.f;
      const float v = XS[(size_t)t * CN + n] * ys;
      XY[(size_t)t * CN + n] = v;
      flat_bf[(((size_t)b * CT + t) * CNH + h) * CN + n] = (bf16_t)v;
    }
  }
}

// ---------------------------------------------------------------------------
// Kernel C: yMLP = flat(4096 x 3072) @ decoder(3072 x 768); 32x32 wave tiles
// ---------------------------------------------------------------------------
__global__ __launch_bounds__(256) void decoder_gemm_kernel(
    const bf16_t* __restrict__ flat_bf, const bf16_t* __restrict__ dec_t,
    float* __restrict__ ymlp) {
  const int lane = threadIdx.x & 31, wave = threadIdx.x >> 5;
  const int wm = wave & 1, wn = wave >> 1;
  const int r0 = blockIdx.x * 64 + wm * 32;
  const int n0 = blockIdx.y * 128 + wn * 32;

  v8f c[2][2];
#pragma unroll
  for (int i = 0; i < 2; ++i)
#pragma unroll
    for (int j = 0; j < 2; ++j)
      c[i][j] = (v8f){0.f,0.f,0.f,0.f,0.f,0.f,0.f,0.f};

  for (int k0 = 0; k0 < CK; k0 += 32) {
    if (k0 + 32 < CK)
      __builtin_prefetch(dec_t + (size_t)n0 * CK + k0 + 32, 0, 1);
    v16bf a0 = load_A_rowmajor(flat_bf, CK, r0,      k0, lane);
    v16bf a1 = load_A_rowmajor(flat_bf, CK, r0 + 16, k0, lane);
    v16bf b0 = load_B_rowsT(dec_t, CK, n0,      k0, lane);
    v16bf b1 = load_B_rowsT(dec_t, CK, n0 + 16, k0, lane);
    c[0][0] = wmma_bf16(a0, b0, c[0][0]);
    c[0][1] = wmma_bf16(a0, b1, c[0][1]);
    c[1][0] = wmma_bf16(a1, b0, c[1][0]);
    c[1][1] = wmma_bf16(a1, b1, c[1][1]);
  }
#pragma unroll
  for (int mi = 0; mi < 2; ++mi) {
#pragma unroll
    for (int ni = 0; ni < 2; ++ni) {
      const int col = n0 + ni * 16 + (lane & 15);
      const int mb  = r0 + mi * 16 + ((lane >> 4) << 3);
#pragma unroll
      for (int i = 0; i < 8; ++i)
        ymlp[(size_t)(mb + i) * CD + col] = c[mi][ni][i];
    }
  }
}

// ---------------------------------------------------------------------------
// Kernel D: out1 = layernorm(x + layernorm(yMLP)*sqrt(reg)*scale)
// ---------------------------------------------------------------------------
__device__ inline float block_sum(float v, float* red) {
  const int tid = threadIdx.x;
  red[tid] = v;
  __syncthreads();
  for (int off = 128; off > 0; off >>= 1) {
    if (tid < off) red[tid] += red[tid + off];
    __syncthreads();
  }
  const float r = red[0];
  __syncthreads();
  return r;
}

__global__ __launch_bounds__(256) void final_ln_kernel(
    const float* __restrict__ x, const float* __restrict__ ymlp,
    const float* __restrict__ scale, const float* __restrict__ ra,
    float* __restrict__ out1) {
  __shared__ float red[256];
  __shared__ float sZ[CD];
  const int row = blockIdx.x;
  const float* yr = ymlp + (size_t)row * CD;
  const float* xr = x    + (size_t)row * CD;
  float* orow = out1 + (size_t)row * CD;
  const int tid = threadIdx.x;

  float s = 0.f, q = 0.f;
  for (int k = tid; k < CD; k += 256) {
    const float v = yr[k];
    s += v; q += v * v;
  }
  const float S1 = block_sum(s, red);
  const float S2 = block_sum(q, red);
  const float m1  = S1 / CD;
  const float rs1 = rsqrtf(S2 / CD - m1 * m1 + 1e-5f);

  float s2 = 0.f, q2 = 0.f;
  for (int k = tid; k < CD; k += 256) {
    const float yl  = (yr[k] - m1) * rs1;
    const float reg = 0.1f / (ra[k] + 1e-6f);
    const float z   = xr[k] + yl * sqrtf(reg) * scale[k];
    sZ[k] = z;
    s2 += z; q2 += z * z;
  }
  const float Z1 = block_sum(s2, red);
  const float Z2 = block_sum(q2, red);
  const float m2  = Z1 / CD;
  const float rs2 = rsqrtf(Z2 / CD - m2 * m2 + 1e-5f);
  __syncthreads();
  for (int k = tid; k < CD; k += 256)
    orow[k] = (sZ[k] - m2) * rs2;
}

// ---------------------------------------------------------------------------
// Launch
// ---------------------------------------------------------------------------
extern "C" void kernel_launch(void* const* d_in, const int* in_sizes, int n_in,
                              void* d_out, int out_size, void* d_ws, size_t ws_size,
                              hipStream_t stream) {
  (void)in_sizes; (void)n_in; (void)out_size; (void)ws_size;
  const float* x     = (const float*)d_in[0]; // B,1,T,D
  const float* enc   = (const float*)d_in[1]; // NH,D,N
  const float* encv  = (const float*)d_in[2]; // NH,D,N
  const float* dec   = (const float*)d_in[3]; // NH*N, D
  const float* scale = (const float*)d_in[4]; // D
  const float* ra    = (const float*)d_in[5]; // D

  float* out1 = (float*)d_out;                 // B*T*D
  float* xy   = out1 + (size_t)CB * CT * CD;   // B*NH*T*N

  char* ws = (char*)d_ws;
  auto alignup = [](size_t v) { return (v + 255) & ~(size_t)255; };
  size_t off = 0;
  bf16_t* xbf     = (bf16_t*)(ws + off); off = alignup(off + (size_t)CB * CT * CD * 2);
  bf16_t* xbf_t   = (bf16_t*)(ws + off); off = alignup(off + (size_t)CB * CD * CT * 2);
  bf16_t* enc_t   = (bf16_t*)(ws + off); off = alignup(off + (size_t)CNH * CN * CD * 2);
  bf16_t* encv_t  = (bf16_t*)(ws + off); off = alignup(off + (size_t)CNH * CN * CD * 2);
  bf16_t* dec_t   = (bf16_t*)(ws + off); off = alignup(off + (size_t)CD * CK * 2);
  float*  xsp     = (float*) (ws + off); off = alignup(off + (size_t)CB * CNH * CT * CN * 4);
  bf16_t* qr      = (bf16_t*)(ws + off); off = alignup(off + (size_t)CB * CNH * CT * CN * 2);
  bf16_t* flat_bf = (bf16_t*)(ws + off); off = alignup(off + (size_t)CB * CT * CK * 2);
  float*  ymlp    = (float*) (ws + off); off = alignup(off + (size_t)CB * CT * CD * 4);

  // stage bf16 operands
  {
    const int nx = CB * CT * CD;
    cvt_f32_bf16<<<(nx + 255) / 256, 256, 0, stream>>>(x, xbf, nx);
    dim3 blk(32, 8);
    transpose_cvt_kernel<<<dim3(CD / 32, CT / 32, CB),  blk, 0, stream>>>(x,    xbf_t,  CT, CD);
    transpose_cvt_kernel<<<dim3(CN / 32, CD / 32, CNH), blk, 0, stream>>>(enc,  enc_t,  CD, CN);
    transpose_cvt_kernel<<<dim3(CN / 32, CD / 32, CNH), blk, 0, stream>>>(encv, encv_t, CD, CN);
    transpose_cvt_kernel<<<dim3(CD / 32, CK / 32, 1),   blk, 0, stream>>>(dec,  dec_t,  CK, CD);
  }

  // x_sparse = relu(x @ encoder)
  encode_gemm_kernel<<<dim3(CT / 64, CN / 128, CB * CNH), 256, 0, stream>>>(
      xbf, enc_t, xsp);

  // QR = rope(x_sparse)
  {
    const int total = CB * CNH * CT * CN;
    rope_kernel<<<total / 256, 256, 0, stream>>>(xsp, qr);
  }

  // fused attention middle -> xy (d_out part 2) + flat_bf
  attn_fused_kernel<<<dim3(CT / 16, CB * CNH), 256, 0, stream>>>(
      qr, xbf_t, xsp, encv_t, xy, flat_bf);

  // yMLP = flat @ decoder
  decoder_gemm_kernel<<<dim3(CB * CT / 64, CD / 128), 256, 0, stream>>>(
      flat_bf, dec_t, ymlp);

  // final layernorms -> out1
  final_ln_kernel<<<CB * CT, 256, 0, stream>>>(x, ymlp, scale, ra, out1);
}
